// Gat_model_21526376087766
// MI455X (gfx1250) — matrix operand order repaired and checked by reading, not compiled, not run on hardware
//
#include <hip/hip_runtime.h>
#include <math.h>

#define N_NODES 50000
#define N_EDGES 600000
#define D 128
#define EDIM 5
#define GR 128       // rows of X per block in the WMMA GEMM
#define LDH2 136     // padded LDS stride in halves: 272B rows = 16B-aligned + 4-bank rotate

typedef __attribute__((ext_vector_type(16))) _Float16 v16h;
typedef __attribute__((ext_vector_type(2)))  _Float16 h2;
typedef __attribute__((ext_vector_type(8)))  float    v8f;
typedef int v4i_ __attribute__((__vector_size__(16)));   // matches async-LDS builtin param

#define AS1 __attribute__((address_space(1)))
#define AS3 __attribute__((address_space(3)))

#if defined(__has_builtin)
#if __has_builtin(__builtin_amdgcn_global_load_async_to_lds_b128) && \
    __has_builtin(__builtin_amdgcn_s_wait_asynccnt)
#define USE_ASYNC_LDS 1
#endif
#endif
#ifndef USE_ASYNC_LDS
#define USE_ASYNC_LDS 0
#endif

#if USE_ASYNC_LDS
__device__ __forceinline__ void async_cp16(const _Float16* gsrc, _Float16* ldst)
{
    // GLOBAL_LOAD_ASYNC_TO_LDS_B128: per-lane 16B memory -> LDS, ASYNCcnt-tracked
    __builtin_amdgcn_global_load_async_to_lds_b128((AS1 v4i_*)gsrc, (AS3 v4i_*)ldst, 0, 0);
}
#endif

// ---------------------------------------------------------------------------
// Pre-pass conversions (run once per launch; trivially bandwidth-bound)
// ---------------------------------------------------------------------------
__global__ void f32_to_f16(const float* __restrict__ in, _Float16* __restrict__ out, int n)
{
    int i = blockIdx.x * blockDim.x + threadIdx.x;
    if (i < n) out[i] = (_Float16)in[i];
}

// Wt[n*128 + k] = (f16) W[k*128 + n]  (transpose so B tiles are contiguous per n)
__global__ void w_to_f16_t(const float* __restrict__ W, _Float16* __restrict__ Wt)
{
    int i = blockIdx.x * blockDim.x + threadIdx.x;   // 16384 threads
    int k = i >> 7, n = i & 127;
    Wt[n * D + k] = (_Float16)W[i];
}

// ---------------------------------------------------------------------------
// WMMA GEMM: out = act( Xh @ W + bias ), K = N = 128, f16 WMMA, f32 accumulate.
// Xh: f16 row-major [nrows x 128]; Wth: f16 transposed [n*128+k].
// Compile-time output config: ACT = leaky(0.01), OUTF = write f32, OUTH = f16.
// Block = 256 threads = 8 waves; wave owns a 16-row stripe, loops 8 col tiles.
// ---------------------------------------------------------------------------
template <bool ACT, bool OUTF, bool OUTH>
__global__ __launch_bounds__(256)
void gemm128_wmma(const _Float16* __restrict__ Xh, const _Float16* __restrict__ Wth,
                  const float* __restrict__ bias, float* __restrict__ Yf,
                  _Float16* __restrict__ Yh, int nrows)
{
    __shared__ _Float16 Bs[D * LDH2];    // Bs[n*LDH2 + k]
    __shared__ _Float16 As[GR * LDH2];   // As[r*LDH2 + k]
    __shared__ float    bs[D];

    const int t = threadIdx.x;
    const int rowBase = blockIdx.x * GR;
    const bool fullBlk = (rowBase + GR) <= nrows;

#if USE_ASYNC_LDS
    // B: 128 rows x 16 chunks of 16B
    for (int c = t; c < D * (D / 8); c += 256) {
        int n = c >> 4, k8 = (c & 15) * 8;
        async_cp16(Wth + n * D + k8, &Bs[n * LDH2 + k8]);
    }
    // A: 128 rows x 16 chunks of 16B (zero-fill ragged rows via plain LDS stores)
    for (int c = t; c < GR * (D / 8); c += 256) {
        int r = c >> 4, k8 = (c & 15) * 8;
        int row = rowBase + r;
        if (row < nrows) {
            async_cp16(Xh + (size_t)row * D + k8, &As[r * LDH2 + k8]);
        } else {
            *(float4*)&As[r * LDH2 + k8] = make_float4(0.f, 0.f, 0.f, 0.f);
        }
    }
    if (t < D) bs[t] = bias[t];
    __builtin_amdgcn_s_wait_asynccnt(0);
#else
    for (int i = t; i < D * D; i += 256) {
        int n = i >> 7, k = i & 127;
        Bs[n * LDH2 + k] = Wth[i];
    }
    for (int i = t; i < GR * D; i += 256) {
        int r = i >> 7, k = i & 127;
        int row = rowBase + r;
        As[r * LDH2 + k] = (row < nrows) ? Xh[(size_t)row * D + k] : (_Float16)0.0f;
    }
    if (t < D) bs[t] = bias[t];
#endif
    __syncthreads();

    const int lane = t & 31;
    const int wave = t >> 5;
    const int hi   = lane >> 4;
    const int m    = lane & 15;

    for (int ct = 0; ct < 8; ++ct) {
        v8f c = {};
#pragma unroll
        for (int k0 = 0; k0 < D; k0 += 32) {
            v16h a, b;
#pragma unroll
            for (int j = 0; j < 8; ++j) {
                // A, contiguous k order used here (same as B): VGPR j holds K=2j,2j+1 (+16 for hi lanes), M=lane&15
                int ka = 16 * hi + 2 * j;
                h2 pa = *(const h2*)&As[(wave * 16 + m) * LDH2 + k0 + ka];
                a[2 * j]     = pa.x;
                a[2 * j + 1] = pa.y;
                // B, contiguous k order used here (same as A): VGPR j holds K=2j,2j+1 (+16 for hi lanes), N=lane&15
                int kb = 16 * hi + 2 * j;
                h2 pb = *(const h2*)&Bs[(ct * 16 + m) * LDH2 + k0 + kb];
                b[2 * j]     = pb.x;
                b[2 * j + 1] = pb.y;
            }
            c = __builtin_amdgcn_wmma_f32_16x16x32_f16(false, a, false, b,
                                                       (short)0, c, false, false);
        }
        // C/D layout: VGPR r -> M = r + 8*hi, N = lane&15
        const int col = ct * 16 + m;
        const float bv = bs[col];
        float vv[8];
#pragma unroll
        for (int r = 0; r < 8; ++r) {
            float v = c[r] + bv;
            if (ACT) v = (v >= 0.0f) ? v : 0.01f * v;
            vv[r] = v;
        }
        const int row0 = rowBase + wave * 16 + 8 * hi;
        const size_t base = (size_t)row0 * D + col;
        if (fullBlk) {
            if (OUTF) {
#pragma unroll
                for (int r = 0; r < 8; ++r) Yf[base + (size_t)r * D] = vv[r];
            }
            if (OUTH) {
#pragma unroll
                for (int r = 0; r < 8; ++r) Yh[base + (size_t)r * D] = (_Float16)vv[r];
            }
        } else {
#pragma unroll
            for (int r = 0; r < 8; ++r) {
                if (row0 + r < nrows) {
                    if (OUTF) Yf[base + (size_t)r * D] = vv[r];
                    if (OUTH) Yh[base + (size_t)r * D] = (_Float16)vv[r];
                }
            }
        }
    }
}

// ---------------------------------------------------------------------------
// LayerNorm (in place on f32) + f16 mirror for downstream WMMA GEMMs.
// ---------------------------------------------------------------------------
__global__ __launch_bounds__(128)
void layernorm_enc(float* __restrict__ Y, const float* __restrict__ g,
                   const float* __restrict__ b, _Float16* __restrict__ Yh)
{
    __shared__ float red[128];
    const int row = blockIdx.x, t = threadIdx.x;
    float v = Y[row * D + t];
    red[t] = v; __syncthreads();
#pragma unroll
    for (int s = 64; s > 0; s >>= 1) { if (t < s) red[t] += red[t + s]; __syncthreads(); }
    float mean = red[0] * (1.0f / D); __syncthreads();
    float d = v - mean;
    red[t] = d * d; __syncthreads();
#pragma unroll
    for (int s = 64; s > 0; s >>= 1) { if (t < s) red[t] += red[t + s]; __syncthreads(); }
    float inv = rsqrtf(red[0] * (1.0f / D) + 1e-5f);
    float o = d * inv * g[t] + b[t];
    Y[row * D + t]  = o;
    Yh[row * D + t] = (_Float16)o;
}

// Residual GAT epilogue: Y += LN(agg + gat_bias)*g + b ; refresh f16 mirror.
__global__ __launch_bounds__(128)
void gat_ln_residual(const float* __restrict__ agg, const float* __restrict__ gb,
                     const float* __restrict__ g, const float* __restrict__ b,
                     float* __restrict__ Y, _Float16* __restrict__ Yh)
{
    __shared__ float red[128];
    const int row = blockIdx.x, t = threadIdx.x;
    float v = agg[row * D + t] + gb[t];
    red[t] = v; __syncthreads();
#pragma unroll
    for (int s = 64; s > 0; s >>= 1) { if (t < s) red[t] += red[t + s]; __syncthreads(); }
    float mean = red[0] * (1.0f / D); __syncthreads();
    float d = v - mean;
    red[t] = d * d; __syncthreads();
#pragma unroll
    for (int s = 64; s > 0; s >>= 1) { if (t < s) red[t] += red[t + s]; __syncthreads(); }
    float inv = rsqrtf(red[0] * (1.0f / D) + 1e-5f);
    float o = Y[row * D + t] + d * inv * g[t] + b[t];
    Y[row * D + t]  = o;
    Yh[row * D + t] = (_Float16)o;
}

// ---------------------------------------------------------------------------
// GAT edge phase
// ---------------------------------------------------------------------------
__global__ void gat_init(float* __restrict__ agg, float* __restrict__ mx,
                         float* __restrict__ den)
{
    int i = blockIdx.x * blockDim.x + threadIdx.x;
    if (i < N_NODES * D) agg[i] = 0.0f;
    if (i < N_NODES) { mx[i] = -__builtin_inff(); den[i] = 0.0f; }
}

__device__ __forceinline__ void atomicMaxF(float* addr, float v)
{
    if (v >= 0.0f) atomicMax((int*)addr, __float_as_int(v));
    else           atomicMin((unsigned int*)addr, __float_as_uint(v));
}

// Wave-per-edge: logits[e] = att . leaky(xl[src] + xr[dst] + ea@We, 0.2);
// segment-max into mx[dst].
__global__ __launch_bounds__(256)
void edge_logits(const float* __restrict__ xl, const float* __restrict__ xr,
                 const float* __restrict__ ea, const int* __restrict__ eidx,
                 const float* __restrict__ We, const float* __restrict__ att,
                 float* __restrict__ logits, float* __restrict__ mx)
{
    __shared__ float WeS[EDIM * D];
    __shared__ float attS[D];
    const int t = threadIdx.x;
    for (int i = t; i < EDIM * D; i += 256) WeS[i] = We[i];
    if (t < D) attS[t] = att[t];
    __syncthreads();

    const int e = blockIdx.x * 8 + (t >> 5);
    if (e >= N_EDGES) return;
    const int lane = t & 31;
    const int s = eidx[e];
    const int d = eidx[N_EDGES + e];
    const float e0 = ea[e * EDIM + 0], e1 = ea[e * EDIM + 1], e2 = ea[e * EDIM + 2],
                e3 = ea[e * EDIM + 3], e4 = ea[e * EDIM + 4];
    float part = 0.0f;
#pragma unroll
    for (int j = 0; j < 4; ++j) {
        int f = j * 32 + lane;
        float v = xl[(size_t)s * D + f] + xr[(size_t)d * D + f]
                + e0 * WeS[0 * D + f] + e1 * WeS[1 * D + f] + e2 * WeS[2 * D + f]
                + e3 * WeS[3 * D + f] + e4 * WeS[4 * D + f];
        v = (v >= 0.0f) ? v : 0.2f * v;
        part += attS[f] * v;
    }
#pragma unroll
    for (int off = 16; off > 0; off >>= 1) part += __shfl_xor(part, off, 32);
    if (lane == 0) { logits[e] = part; atomicMaxF(&mx[d], part); }
}

__global__ void edge_exp(float* __restrict__ logits, const int* __restrict__ eidx,
                         const float* __restrict__ mx, float* __restrict__ den)
{
    int e = blockIdx.x * blockDim.x + threadIdx.x;
    if (e >= N_EDGES) return;
    int d = eidx[N_EDGES + e];
    float ex = __expf(logits[e] - mx[d]);
    logits[e] = ex;
    atomicAdd(&den[d], ex);
}

__global__ __launch_bounds__(256)
void edge_aggregate(const float* __restrict__ xl, const int* __restrict__ eidx,
                    const float* __restrict__ ex, const float* __restrict__ den,
                    float* __restrict__ agg)
{
    const int t = threadIdx.x;
    const int e = blockIdx.x * 8 + (t >> 5);
    if (e >= N_EDGES) return;
    const int lane = t & 31;
    const int s = eidx[e];
    const int d = eidx[N_EDGES + e];
    const float alpha = ex[e] / (den[d] + 1e-16f);
#pragma unroll
    for (int j = 0; j < 4; ++j) {
        int f = j * 32 + lane;
        atomicAdd(&agg[(size_t)d * D + f], alpha * xl[(size_t)s * D + f]);
    }
}

// ---------------------------------------------------------------------------
// Decoder: small dense layers with W cached in LDS (odd dims 87/46/4)
// ---------------------------------------------------------------------------
template <int KD, int ND>
__global__ __launch_bounds__(256)
void dense_small(const float* __restrict__ X, const float* __restrict__ W,
                 const float* __restrict__ bias, float* __restrict__ Y,
                 int nrows, int act)
{
    __shared__ float Ws[KD * ND];
    __shared__ float bs[ND];
    const int t = threadIdx.x;
    for (int i = t; i < KD * ND; i += 256) Ws[i] = W[i];
    for (int i = t; i < ND; i += 256) bs[i] = bias[i];
    __syncthreads();

    const int total = nrows * ND;
    for (int idx = blockIdx.x * 256 + t; idx < total; idx += gridDim.x * 256) {
        int r = idx / ND, n = idx % ND;
        const float* xr = X + (size_t)r * KD;
        float acc = bs[n];
#pragma unroll 8
        for (int k = 0; k < KD; ++k) acc += xr[k] * Ws[k * ND + n];
        if (act) acc = (acc >= 0.0f) ? acc : 0.01f * acc;
        Y[idx] = acc;
    }
}

// ---------------------------------------------------------------------------
extern "C" void kernel_launch(void* const* d_in, const int* in_sizes, int n_in,
                              void* d_out, int out_size, void* d_ws, size_t ws_size,
                              hipStream_t stream)
{
    const float* x        = (const float*)d_in[0];
    const int*   eidx     = (const int*)  d_in[1];
    const float* ea       = (const float*)d_in[2];
    const float* enc_W1   = (const float*)d_in[3];
    const float* enc_b1   = (const float*)d_in[4];
    const float* enc_W2   = (const float*)d_in[5];
    const float* enc_b2   = (const float*)d_in[6];
    const float* enc_W3   = (const float*)d_in[7];
    const float* enc_b3   = (const float*)d_in[8];
    const float* enc_ln_g = (const float*)d_in[9];
    const float* enc_ln_b = (const float*)d_in[10];
    const float* gat_Wl   = (const float*)d_in[11];
    const float* gat_bl   = (const float*)d_in[12];
    const float* gat_Wr   = (const float*)d_in[13];
    const float* gat_br   = (const float*)d_in[14];
    const float* gat_We   = (const float*)d_in[15];
    const float* gat_att  = (const float*)d_in[16];
    const float* gat_bias = (const float*)d_in[17];
    const float* ln_g     = (const float*)d_in[18];
    const float* ln_b     = (const float*)d_in[19];
    const float* dec_W1   = (const float*)d_in[20];
    const float* dec_b1   = (const float*)d_in[21];
    const float* dec_W2   = (const float*)d_in[22];
    const float* dec_b2   = (const float*)d_in[23];
    const float* dec_W3   = (const float*)d_in[24];
    const float* dec_b3   = (const float*)d_in[25];
    float* out = (float*)d_out;

    // -------- workspace carve-up --------
    const size_t ND_ = (size_t)N_NODES * D;
    float* ws = (float*)d_ws;
    size_t off = 0;
    float* y   = ws + off; off += ND_;          // node features (f32)
    float* t0  = ws + off; off += ND_;          // xl / dec1
    float* t1  = ws + off; off += ND_;          // xr / dec2
    float* agg = ws + off; off += ND_;          // GAT aggregation
    float* lg  = ws + off; off += (size_t)N_EDGES;
    float* mx  = ws + off; off += (size_t)N_NODES;
    float* den = ws + off; off += (size_t)N_NODES;
    // f16 region
    _Float16* hbase = (_Float16*)(ws + off);
    size_t hoff = 0;
    _Float16* xh  = hbase + hoff; hoff += ND_;  // f16 input
    _Float16* h0  = hbase + hoff; hoff += ND_;
    _Float16* h1  = hbase + hoff; hoff += ND_;
    _Float16* yh  = hbase + hoff; hoff += ND_;  // f16 mirror of y
    _Float16* wth = hbase + hoff;               // 9 transposed f16 weight mats
    const size_t WSZ = (size_t)D * D;

    const int gemmGrid  = (N_NODES + GR - 1) / GR;       // 391
    const int eWaveGrid = (N_EDGES + 7) / 8;
    const int eThrGrid  = (N_EDGES + 255) / 256;
    const int initGrid  = (N_NODES * D + 255) / 256;
    const int cvtGrid   = (int)((ND_ + 255) / 256);
    const int wGrid     = (int)(WSZ / 256);

    // -------- pre-pass: f16 conversions + weight transposes --------
    f32_to_f16<<<cvtGrid, 256, 0, stream>>>(x, xh, (int)ND_);
    w_to_f16_t<<<wGrid, 256, 0, stream>>>(enc_W1, wth + 0 * WSZ);
    w_to_f16_t<<<wGrid, 256, 0, stream>>>(enc_W2, wth + 1 * WSZ);
    w_to_f16_t<<<wGrid, 256, 0, stream>>>(enc_W3, wth + 2 * WSZ);
    for (int i = 0; i < 3; ++i) {
        w_to_f16_t<<<wGrid, 256, 0, stream>>>(gat_Wl + i * WSZ, wth + (3 + i) * WSZ);
        w_to_f16_t<<<wGrid, 256, 0, stream>>>(gat_Wr + i * WSZ, wth + (6 + i) * WSZ);
    }

    // -------- encoder MLP + LN --------
    gemm128_wmma<true,  false, true><<<gemmGrid, 256, 0, stream>>>(
        xh, wth + 0 * WSZ, enc_b1, nullptr, h0, N_NODES);
    gemm128_wmma<true,  false, true><<<gemmGrid, 256, 0, stream>>>(
        h0, wth + 1 * WSZ, enc_b2, nullptr, h1, N_NODES);
    gemm128_wmma<false, true, false><<<gemmGrid, 256, 0, stream>>>(
        h1, wth + 2 * WSZ, enc_b3, y, nullptr, N_NODES);
    layernorm_enc<<<N_NODES, 128, 0, stream>>>(y, enc_ln_g, enc_ln_b, yh);

    // -------- 3x GATv2 --------
    for (int i = 0; i < 3; ++i) {
        const float* bl   = gat_bl   + (size_t)i * D;
        const float* br   = gat_br   + (size_t)i * D;
        const float* We   = gat_We   + (size_t)i * EDIM * D;
        const float* at   = gat_att  + (size_t)i * D;
        const float* gb   = gat_bias + (size_t)i * D;
        const float* lg_g = ln_g     + (size_t)i * D;
        const float* lg_b = ln_b     + (size_t)i * D;

        gemm128_wmma<false, true, false><<<gemmGrid, 256, 0, stream>>>(
            yh, wth + (3 + i) * WSZ, bl, t0, nullptr, N_NODES);   // xl
        gemm128_wmma<false, true, false><<<gemmGrid, 256, 0, stream>>>(
            yh, wth + (6 + i) * WSZ, br, t1, nullptr, N_NODES);   // xr
        gat_init<<<initGrid, 256, 0, stream>>>(agg, mx, den);
        edge_logits<<<eWaveGrid, 256, 0, stream>>>(t0, t1, ea, eidx, We, at, lg, mx);
        edge_exp<<<eThrGrid, 256, 0, stream>>>(lg, eidx, mx, den);
        edge_aggregate<<<eWaveGrid, 256, 0, stream>>>(t0, eidx, lg, den, agg);
        gat_ln_residual<<<N_NODES, 128, 0, stream>>>(agg, gb, lg_g, lg_b, y, yh);
    }

    // -------- decoder MLP --------
    dense_small<128, 87><<<(N_NODES * 87 + 255) / 256, 256, 0, stream>>>(
        y, dec_W1, dec_b1, t0, N_NODES, 1);
    dense_small<87, 46><<<(N_NODES * 46 + 255) / 256, 256, 0, stream>>>(
        t0, dec_W2, dec_b2, t1, N_NODES, 1);
    dense_small<46, 4><<<(N_NODES * 4 + 255) / 256, 256, 0, stream>>>(
        t1, dec_W3, dec_b3, out, N_NODES, 0);
}
